// MultiHeadAttention_63161789055036
// MI455X (gfx1250) — compile-verified, matrix operand order checked by reading, not run on hardware
//
#include <hip/hip_runtime.h>
#include <hip/hip_bf16.h>

// ---------------------------------------------------------------------------
// Causal multi-head attention forward for MI455X (gfx1250), wave32 + WMMA.
// All GEMM-shaped math goes through v_wmma_f32_16x16x32_f16 (f32 accumulate).
// ---------------------------------------------------------------------------

typedef _Float16 h4   __attribute__((ext_vector_type(4)));
typedef _Float16 h8   __attribute__((ext_vector_type(8)));
typedef _Float16 v16h __attribute__((ext_vector_type(16)));
typedef float    v8f  __attribute__((ext_vector_type(8)));
typedef float    f4   __attribute__((ext_vector_type(4)));

#define S_LEN   2048
#define D_MODEL 1024
#define NHEADS  16
#define DHEAD   64
#define BATCH   2
#define MROWS   (BATCH * S_LEN)   // 4096

// -------------------------------------------------------------------------
// 16x32 f16 fragment load (ISA 7.12.2 layout):
//   lane L holds row (L & 15); kbase = (L >> 4) * 8
//   vector elems 0..7  -> K = kbase .. kbase+7
//   vector elems 8..15 -> K = kbase+16 .. kbase+23
// Works for A fragments (row = M) and B fragments (row = N column of B),
// for any row-major source with given stride. Two 16-byte loads per lane.
// -------------------------------------------------------------------------
__device__ __forceinline__ v16h load_frag(const _Float16* base, int stride, int lane) {
  const _Float16* p = base + (size_t)(lane & 15) * stride + ((lane >> 4) << 3);
  h8 lo = *(const h8*)(p);
  h8 hi = *(const h8*)(p + 16);
  v16h f;
#pragma unroll
  for (int i = 0; i < 8; ++i) { f[i] = lo[i]; f[i + 8] = hi[i]; }
  return f;
}

__device__ __forceinline__ v8f vzero8() {
  v8f z;
#pragma unroll
  for (int i = 0; i < 8; ++i) z[i] = 0.0f;
  return z;
}

// -------------------------------------------------------------------------
// WMMA GEMM:  C[M=4096, N=1024] = A[4096,1024] * W[1024,1024] + bias
//   AF32    : 1 -> A is fp32 (x), convert to f16 in LDS; 0 -> A already f16
//   OUTMODE : 0 -> fp32 [M,N] (final output)
//             1 -> f16 head layout   [BH][S][Dh]   (Q, K)
//             2 -> f16 head-T layout [BH][Dh][S]   (V transposed)
// Block tile 256x64, BK=32, 256 threads = 8 waves; wave w owns a 32x64 slab
// (2 A-frags x 4 B-frags -> 8 WMMAs per K-step, B-frags reused across rows).
// -------------------------------------------------------------------------
template<int AF32, int OUTMODE>
__global__ __launch_bounds__(256) void gemm_wmma_kernel(
    const void* __restrict__ Aptr, const float* __restrict__ W,
    const float* __restrict__ bias, void* __restrict__ outp)
{
  constexpr int BM = 256, BN = 64, BK = 32;
  constexpr int LDT = 40;                    // padded LDS stride (halfs)
  __shared__ _Float16 As[BM * LDT];
  __shared__ _Float16 Bs[BN * LDT];

  const int t    = threadIdx.x;
  const int lane = t & 31;
  const int wave = t >> 5;                   // 0..7
  const int n0   = blockIdx.x * BN;
  const int m0   = blockIdx.y * BM;

  v8f acc[2][4];
#pragma unroll
  for (int rt = 0; rt < 2; ++rt)
#pragma unroll
    for (int ns = 0; ns < 4; ++ns) acc[rt][ns] = vzero8();

  for (int k0 = 0; k0 < D_MODEL; k0 += BK) {
    const bool more = (k0 + BK) < D_MODEL;
    // ---- stage A tile (256x32) into LDS as f16 -------------------------
#pragma unroll
    for (int j = 0; j < 8; ++j) {
      int idx = j * 256 + t;                 // 0..2047 4-elem chunks
      int row = idx >> 3;                    // 8 chunks per 32-wide row
      int col = (idx & 7) << 2;
      h4 hv;
      if (AF32) {
        const float* A = (const float*)Aptr;
        const float* gp = A + (size_t)(m0 + row) * D_MODEL + k0 + col;
        if (more) __builtin_prefetch(gp + BK, 0, 1);   // global_prefetch_b8
        f4 v = *(const f4*)gp;
#pragma unroll
        for (int c = 0; c < 4; ++c) hv[c] = (_Float16)v[c];
      } else {
        const _Float16* A = (const _Float16*)Aptr;
        const _Float16* gp = A + (size_t)(m0 + row) * D_MODEL + k0 + col;
        if (more) __builtin_prefetch(gp + BK, 0, 1);
        hv = *(const h4*)gp;
      }
      *(h4*)(&As[row * LDT + col]) = hv;
    }
    // ---- stage W tile (32x64) transposed into LDS: Bs[n][k] ------------
#pragma unroll
    for (int j = 0; j < 2; ++j) {
      int idx = j * 256 + t;                 // 0..511
      int kr  = idx >> 4;                    // 0..31
      int nc  = (idx & 15) << 2;             // 0..60
      const float* gp = W + (size_t)(k0 + kr) * D_MODEL + n0 + nc;
      if (more) __builtin_prefetch(gp + (size_t)BK * D_MODEL, 0, 1);
      f4 v = *(const f4*)gp;
#pragma unroll
      for (int c = 0; c < 4; ++c)
        Bs[(nc + c) * LDT + kr] = (_Float16)v[c];
    }
    __syncthreads();

    v16h af0 = load_frag(&As[(wave * 32) * LDT],      LDT, lane);
    v16h af1 = load_frag(&As[(wave * 32 + 16) * LDT], LDT, lane);
#pragma unroll
    for (int ns = 0; ns < 4; ++ns) {
      v16h bfrag = load_frag(&Bs[(ns * 16) * LDT], LDT, lane);
      acc[0][ns] = __builtin_amdgcn_wmma_f32_16x16x32_f16(
          false, af0, false, bfrag, (short)0, acc[0][ns], false, false);
      acc[1][ns] = __builtin_amdgcn_wmma_f32_16x16x32_f16(
          false, af1, false, bfrag, (short)0, acc[1][ns], false, false);
    }
    __syncthreads();
  }

  // ---- epilogue: bias + store --------------------------------------------
#pragma unroll
  for (int rt = 0; rt < 2; ++rt) {
#pragma unroll
    for (int ns = 0; ns < 4; ++ns) {
      int gc = n0 + ns * 16 + (lane & 15);
      float bv = bias[gc];
#pragma unroll
      for (int r = 0; r < 8; ++r) {
        int gr = m0 + wave * 32 + rt * 16 + r + ((lane >> 4) << 3);
        float v = acc[rt][ns][r] + bv;
        if (OUTMODE == 0) {
          ((float*)outp)[(size_t)gr * D_MODEL + gc] = v;
        } else {
          int bb = gr >> 11, s = gr & (S_LEN - 1);
          int hh = gc >> 6,  dh = gc & (DHEAD - 1);
          size_t bh = (size_t)bb * NHEADS + hh;
          _Float16* oh = (_Float16*)outp;
          if (OUTMODE == 1) oh[(bh * S_LEN + s) * DHEAD + dh] = (_Float16)v;
          else              oh[(bh * DHEAD + dh) * S_LEN + s] = (_Float16)v;
        }
      }
    }
  }
}

// -------------------------------------------------------------------------
// Flash-style causal attention. Grid (BH=32, S/64=32), 128 threads = 4 waves;
// each wave owns one 16-row query tile, iterates key blocks of 32.
// Scores: 4 WMMAs / block (Dh=64 => two x32 K-steps per 16-key tile).
// P (16x32) is staged via per-wave LDS to convert C-layout -> A-layout,
// then PV: 4 WMMAs / block into f32 accumulators. Mask applied before the
// 1/sqrt(Dh) scale, matching the reference ordering.
// -------------------------------------------------------------------------
__global__ __launch_bounds__(128) void attn_kernel(
    const _Float16* __restrict__ Qh, const _Float16* __restrict__ Kh,
    const _Float16* __restrict__ Vt, _Float16* __restrict__ ctx)
{
  __shared__ _Float16 Pb[4][16 * 32];
  const int lane = threadIdx.x & 31;
  const int wave = threadIdx.x >> 5;
  const int bh   = blockIdx.x;
  const int b    = bh >> 4, h = bh & (NHEADS - 1);
  const int q0   = blockIdx.y * 64 + wave * 16;

  const _Float16* Qbase = Qh + (size_t)bh * S_LEN * DHEAD;
  const _Float16* Kbase = Kh + (size_t)bh * S_LEN * DHEAD;
  const _Float16* Vbase = Vt + (size_t)bh * DHEAD * S_LEN;

  v16h qf0 = load_frag(Qbase + (size_t)q0 * DHEAD,      DHEAD, lane); // K 0..31
  v16h qf1 = load_frag(Qbase + (size_t)q0 * DHEAD + 32, DHEAD, lane); // K 32..63

  v8f o[4];
#pragma unroll
  for (int ns = 0; ns < 4; ++ns) o[ns] = vzero8();
  float mrow[8], lrow[8];
#pragma unroll
  for (int r = 0; r < 8; ++r) { mrow[r] = -1e30f; lrow[r] = 0.0f; }

  const int rowoff = (lane >> 4) << 3;     // 0 or 8
  const int col_l  = lane & 15;
  const int nblocks = (q0 + 15) / 32 + 1;  // keys 0 .. q0+15 in blocks of 32
  _Float16* P = Pb[wave];

  for (int jb = 0; jb < nblocks; ++jb) {
    const int j = jb * 32;
    if (jb + 1 < nblocks) {                // prefetch next key/value block
      __builtin_prefetch(Kbase + (size_t)(j + 32 + (lane & 31)) * DHEAD, 0, 1);
      __builtin_prefetch(Vbase + (size_t)(lane << 1) * S_LEN + j + 32, 0, 1);
    }
    // scores for keys j..j+15 (s0) and j+16..j+31 (s1)
    v16h k00 = load_frag(Kbase + (size_t)j * DHEAD,             DHEAD, lane);
    v16h k01 = load_frag(Kbase + (size_t)j * DHEAD + 32,        DHEAD, lane);
    v16h k10 = load_frag(Kbase + (size_t)(j + 16) * DHEAD,      DHEAD, lane);
    v16h k11 = load_frag(Kbase + (size_t)(j + 16) * DHEAD + 32, DHEAD, lane);

    v8f s0 = vzero8();
    s0 = __builtin_amdgcn_wmma_f32_16x16x32_f16(false, qf0, false, k00, (short)0, s0, false, false);
    s0 = __builtin_amdgcn_wmma_f32_16x16x32_f16(false, qf1, false, k01, (short)0, s0, false, false);
    v8f s1 = vzero8();
    s1 = __builtin_amdgcn_wmma_f32_16x16x32_f16(false, qf0, false, k10, (short)0, s1, false, false);
    s1 = __builtin_amdgcn_wmma_f32_16x16x32_f16(false, qf1, false, k11, (short)0, s1, false, false);

    const bool full = (j + 31) <= q0;      // block entirely below diagonal

#pragma unroll
    for (int r = 0; r < 8; ++r) {
      const int row = q0 + r + rowoff;
      float a = s0[r], c = s1[r];
      if (full) { a *= 0.125f; c *= 0.125f; }
      else {
        int c0 = j + col_l, c1 = c0 + 16;
        a = (c0 <= row) ? a * 0.125f : -1e30f;
        c = (c1 <= row) ? c * 0.125f : -1e30f;
      }
      // row max across the 16 lanes holding this row
      float t = fmaxf(a, c);
#pragma unroll
      for (int off = 1; off < 16; off <<= 1)
        t = fmaxf(t, __shfl_xor(t, off, 32));
      float mnew = fmaxf(mrow[r], t);
      float corr = __expf(mrow[r] - mnew);
      float p0 = __expf(a - mnew);
      float p1 = __expf(c - mnew);
      float rs = p0 + p1;
#pragma unroll
      for (int off = 1; off < 16; off <<= 1)
        rs += __shfl_xor(rs, off, 32);
      mrow[r] = mnew;
      lrow[r] = lrow[r] * corr + rs;
#pragma unroll
      for (int ns = 0; ns < 4; ++ns) o[ns][r] *= corr;
      // C-layout -> row-major P tile in LDS (per-wave private buffer)
      const int prow = r + rowoff;
      P[prow * 32 + col_l]      = (_Float16)p0;
      P[prow * 32 + 16 + col_l] = (_Float16)p1;
    }
    asm volatile("s_wait_dscnt 0" ::: "memory");    // wave-local LDS RAW fence

    v16h pf = load_frag(P, 32, lane);               // P as 16x32 A fragment
#pragma unroll
    for (int ns = 0; ns < 4; ++ns) {
      // B frag: V^T rows are head-dims, contiguous along keys
      v16h vf = load_frag(Vbase + (size_t)(ns * 16) * S_LEN + j, S_LEN, lane);
      o[ns] = __builtin_amdgcn_wmma_f32_16x16x32_f16(
          false, pf, false, vf, (short)0, o[ns], false, false);
    }
  }

  // normalize and scatter ctx as f16 [B][S][D] for the output projection
#pragma unroll
  for (int r = 0; r < 8; ++r) {
    const int row = q0 + r + rowoff;
    const float inv = 1.0f / lrow[r];
#pragma unroll
    for (int ns = 0; ns < 4; ++ns) {
      float v = o[ns][r] * inv;
      int dcol = h * DHEAD + ns * 16 + col_l;
      ctx[((size_t)(b * S_LEN + row)) * D_MODEL + dcol] = (_Float16)v;
    }
  }
}

// -------------------------------------------------------------------------
extern "C" void kernel_launch(void* const* d_in, const int* in_sizes, int n_in,
                              void* d_out, int out_size, void* d_ws, size_t ws_size,
                              hipStream_t stream) {
  (void)in_sizes; (void)n_in; (void)out_size; (void)ws_size;
  const float* x  = (const float*)d_in[0];
  const float* Wq = (const float*)d_in[1];
  const float* bq = (const float*)d_in[2];
  const float* Wk = (const float*)d_in[3];
  const float* bk = (const float*)d_in[4];
  const float* Wv = (const float*)d_in[5];
  const float* bv = (const float*)d_in[6];
  const float* Wo = (const float*)d_in[7];
  const float* bo = (const float*)d_in[8];
  float* out = (float*)d_out;

  char* ws = (char*)d_ws;
  const size_t seg = (size_t)BATCH * NHEADS * S_LEN * DHEAD * sizeof(_Float16); // 8 MiB
  _Float16* Qh  = (_Float16*)(ws);
  _Float16* Kh  = (_Float16*)(ws + seg);
  _Float16* Vt  = (_Float16*)(ws + 2 * seg);
  _Float16* ctx = (_Float16*)(ws + 3 * seg);

  dim3 gg(D_MODEL / 64, MROWS / 256);  // (16, 16)
  gemm_wmma_kernel<1, 1><<<gg, 256, 0, stream>>>(x,   Wq, bq, Qh);
  gemm_wmma_kernel<1, 1><<<gg, 256, 0, stream>>>(x,   Wk, bk, Kh);
  gemm_wmma_kernel<1, 2><<<gg, 256, 0, stream>>>(x,   Wv, bv, Vt);
  attn_kernel<<<dim3(BATCH * NHEADS, S_LEN / 64), 128, 0, stream>>>(Qh, Kh, Vt, ctx);
  gemm_wmma_kernel<0, 0><<<gg, 256, 0, stream>>>(ctx, Wo, bo, out);
}
